// CrossAttention_75625784148331
// MI455X (gfx1250) — matrix operand order repaired and checked by reading, not compiled, not run on hardware
//
#include <hip/hip_runtime.h>
#include <hip/hip_bf16.h>

typedef __bf16 bf16;
typedef __attribute__((ext_vector_type(16))) __bf16 v16bf;
typedef __attribute__((ext_vector_type(8)))  __bf16 v8bf;
typedef __attribute__((ext_vector_type(8)))  float  v8f;
typedef int v4i __attribute__((vector_size(16)));

#define NEGC (-1000000.0f)

// ---------------------------------------------------------------------------
// CDNA5 async global->LDS copy (ASYNCcnt path). Builtin signature (from the
// toolchain's own diagnostic): (AS1 int4*, AS3 int4*, imm offset, imm cpol).
// ---------------------------------------------------------------------------
#if defined(__has_builtin)
#if __has_builtin(__builtin_amdgcn_global_load_async_to_lds_b128) && \
    __has_builtin(__builtin_amdgcn_s_wait_asynccnt)
#define HAVE_ASYNC_LDS 1
#endif
#endif
#ifndef HAVE_ASYNC_LDS
#define HAVE_ASYNC_LDS 0
#endif

#if HAVE_ASYNC_LDS
typedef __attribute__((address_space(1))) v4i* gv4i_ptr;
typedef __attribute__((address_space(3))) v4i* lv4i_ptr;
__device__ __forceinline__ void async_cp16(bf16* lds_dst, const bf16* gsrc) {
    __builtin_amdgcn_global_load_async_to_lds_b128((gv4i_ptr)(gsrc),
                                                   (lv4i_ptr)(lds_dst), 0, 0);
}
__device__ __forceinline__ void async_wait0() {
    __builtin_amdgcn_s_wait_asynccnt(0);
}
#else
__device__ __forceinline__ void async_wait0() {}
#endif

// ---------------------------------------------------------------------------
// Fragment helpers (CDNA5 WMMA 16x16x32 bf16 layouts, wave32)
// ---------------------------------------------------------------------------
__device__ __forceinline__ v16bf load_frag_a(const bf16* base, int ld, int lane) {
    const int hlf = lane >> 4, r = lane & 15;
    const bf16* row = base + (size_t)r * ld + hlf * 8;
    v8bf lo = *(const v8bf*)(row);        // K = hlf*8 .. +7
    v8bf hi = *(const v8bf*)(row + 16);   // K = 16 + hlf*8 .. +7
    v16bf a;
#pragma unroll
    for (int i = 0; i < 8; ++i) { a[i] = lo[i]; a[8 + i] = hi[i]; }
    return a;
}

__device__ __forceinline__ v16bf load_frag_b(const bf16* w, int ld, int lane) {
    const int khalf = lane >> 4, n = lane & 15;
    const bf16* p = w + (size_t)n * ld + khalf * 8;
    v8bf lo = *(const v8bf*)(p);
    v8bf hi = *(const v8bf*)(p + 16);
    v16bf b;
#pragma unroll
    for (int i = 0; i < 8; ++i) { b[i] = lo[i]; b[8 + i] = hi[i]; }
    return b;
}

__device__ __forceinline__ v8f wmma_bf16(v16bf a, v16bf b, v8f c) {
    return __builtin_amdgcn_wmma_f32_16x16x32_bf16(false, a, false, b, (short)0, c,
                                                   false, false);
}

// ---------------------------------------------------------------------------
// Prep kernels
// ---------------------------------------------------------------------------
__global__ __launch_bounds__(256) void add_pe_kernel(const float* __restrict__ x,
                                                     const float* __restrict__ pe,
                                                     bf16* __restrict__ out,
                                                     int LD, int total4) {
    int i = blockIdx.x * 256 + threadIdx.x;
    if (i >= total4) return;
    size_t base = (size_t)i * 4;
    float4 xv = *(const float4*)(x + base);
    float4 pv = *(const float4*)(pe + (base % (size_t)LD));
    out[base + 0] = (bf16)(xv.x + pv.x);
    out[base + 1] = (bf16)(xv.y + pv.y);
    out[base + 2] = (bf16)(xv.z + pv.z);
    out[base + 3] = (bf16)(xv.w + pv.w);
}

__global__ __launch_bounds__(256) void cvt_bf16_kernel(const float* __restrict__ x,
                                                       bf16* __restrict__ out, int total4) {
    int i = blockIdx.x * 256 + threadIdx.x;
    if (i >= total4) return;
    size_t base = (size_t)i * 4;
    float4 xv = *(const float4*)(x + base);
    out[base + 0] = (bf16)xv.x;
    out[base + 1] = (bf16)xv.y;
    out[base + 2] = (bf16)xv.z;
    out[base + 3] = (bf16)xv.w;
}

// ---------------------------------------------------------------------------
// GEMM: C[m,n] = alpha * (sum_k A[m,k]*W[n,k] + bias[n])
// 128x64 block tile, 8 waves x 32x32, double-buffered 64-wide K slabs.
// mode 0: bf16 -> (B,16,Lrows,64); mode 1: bf16 -> (B,16,64,Lrows);
// mode 2: f32 row-major.
// ---------------------------------------------------------------------------
#define GBM 128
#define GBN 64
#define GBK 64
#define GLD 80   // padded LDS row stride (halfs); 160B keeps 16B alignment

__global__ __launch_bounds__(256) void gemm_bf16_kernel(
    const bf16* __restrict__ A, const bf16* __restrict__ W,
    const float* __restrict__ bias, void* __restrict__ dst,
    int M, int N, int K, int mode, int Lrows, float alpha) {
    __shared__ __attribute__((aligned(16))) bf16 As[2][GBM][GLD];
    __shared__ __attribute__((aligned(16))) bf16 Ws[2][GBN][GLD];

    const int tid = threadIdx.x;
    const int lane = tid & 31, wave = tid >> 5;
    const int m0 = blockIdx.x * GBM, n0 = blockIdx.y * GBN;
    const int wm = (wave & 3) * 32, wn = (wave >> 2) * 32;
    const int lr = tid >> 3;           // 0..31
    const int lc = (tid & 7) * 8;      // 0..56

    // stage one 64-wide K slab (A: 128 rows, W: 64 rows) into buffer `buf`
    auto stage = [&](int buf, int k0) {
#if HAVE_ASYNC_LDS
#pragma unroll
        for (int it = 0; it < 4; ++it) {
            int r = lr + it * 32;
            async_cp16(&As[buf][r][lc], A + (size_t)(m0 + r) * K + k0 + lc);
        }
#pragma unroll
        for (int it = 0; it < 2; ++it) {
            int r = lr + it * 32;
            async_cp16(&Ws[buf][r][lc], W + (size_t)(n0 + r) * K + k0 + lc);
        }
#else
        v8bf ta[4], tw[2];
#pragma unroll
        for (int it = 0; it < 4; ++it)
            ta[it] = *(const v8bf*)(A + (size_t)(m0 + lr + it * 32) * K + k0 + lc);
#pragma unroll
        for (int it = 0; it < 2; ++it)
            tw[it] = *(const v8bf*)(W + (size_t)(n0 + lr + it * 32) * K + k0 + lc);
#pragma unroll
        for (int it = 0; it < 4; ++it) *(v8bf*)(&As[buf][lr + it * 32][lc]) = ta[it];
#pragma unroll
        for (int it = 0; it < 2; ++it) *(v8bf*)(&Ws[buf][lr + it * 32][lc]) = tw[it];
#endif
    };

    v8f c[2][2] = {};

    stage(0, 0);
    async_wait0();
    __syncthreads();

    const int nsteps = K / GBK;
    for (int s = 0; s < nsteps; ++s) {
        const int buf = s & 1;
        if (s + 1 < nsteps) stage(buf ^ 1, (s + 1) * GBK);
#pragma unroll
        for (int kk = 0; kk < GBK; kk += 32) {
            v16bf a0 = load_frag_a(&As[buf][wm][kk], GLD, lane);
            v16bf a1 = load_frag_a(&As[buf][wm + 16][kk], GLD, lane);
            v16bf b0 = load_frag_b(&Ws[buf][wn][kk], GLD, lane);
            v16bf b1 = load_frag_b(&Ws[buf][wn + 16][kk], GLD, lane);
            c[0][0] = wmma_bf16(a0, b0, c[0][0]);
            c[0][1] = wmma_bf16(a0, b1, c[0][1]);
            c[1][0] = wmma_bf16(a1, b0, c[1][0]);
            c[1][1] = wmma_bf16(a1, b1, c[1][1]);
        }
        async_wait0();
        __syncthreads();
    }

    const int hlf = lane >> 4, r = lane & 15;
#pragma unroll
    for (int mi = 0; mi < 2; ++mi)
#pragma unroll
        for (int ni = 0; ni < 2; ++ni)
#pragma unroll
            for (int j = 0; j < 8; ++j) {
                int m = m0 + wm + mi * 16 + j + hlf * 8;
                int n = n0 + wn + ni * 16 + r;
                float val = (c[mi][ni][j] + bias[n]) * alpha;
                if (mode == 0) {
                    int b = m / Lrows, rp = m - b * Lrows;
                    int h = n >> 6, d = n & 63;
                    ((bf16*)dst)[(((size_t)(b * 16 + h) * Lrows + rp) << 6) + d] =
                        (bf16)val;
                } else if (mode == 1) {
                    int b = m / Lrows, rp = m - b * Lrows;
                    int h = n >> 6, d = n & 63;
                    ((bf16*)dst)[((size_t)(b * 16 + h) * 64 + d) * Lrows + rp] =
                        (bf16)val;
                } else {
                    ((float*)dst)[(size_t)m * N + n] = val;
                }
            }
}

// ---------------------------------------------------------------------------
// Flash attention: grid (B*H, Lq/128), 256 threads (8 waves x 16 q-rows).
// Q pre-scaled by 1/sqrt(64). Double-buffered 64-column K/V chunks.
// ---------------------------------------------------------------------------
#define ALQ 512
#define ALK 4096
#define AKC 64

__global__ __launch_bounds__(256) void attn_kernel(
    const bf16* __restrict__ Q,   // (B,H,Lq,64)
    const bf16* __restrict__ Kb,  // (B,H,Lk,64)
    const bf16* __restrict__ Vt,  // (B,H,64,Lk)
    const int* __restrict__ qmask,  // (B,Lq)
    const int* __restrict__ fmask,  // (B,Lk)
    bf16* __restrict__ O) {         // (B,Lq,1024), head h at col h*64
    __shared__ __attribute__((aligned(16))) bf16 Ks[2][AKC][GLD];
    __shared__ __attribute__((aligned(16))) bf16 Vs[2][64][GLD];
    __shared__ __attribute__((aligned(16))) bf16 Ps[8][16][GLD];

    const int bh = blockIdx.x;
    const int b = bh >> 4, h = bh & 15;
    const int qbase = blockIdx.y * 128;
    const int tid = threadIdx.x, lane = tid & 31, wave = tid >> 5;
    const int qrow0 = qbase + wave * 16;
    const int hlf = lane >> 4, r = lane & 15;
    const int lr = tid >> 3, lc = (tid & 7) * 8;

    auto stageKV = [&](int buf, int kt) {
#if HAVE_ASYNC_LDS
#pragma unroll
        for (int it = 0; it < 2; ++it) {
            int rr = lr + it * 32;
            async_cp16(&Ks[buf][rr][lc],
                       Kb + ((size_t)bh * ALK + kt + rr) * 64 + lc);
        }
#pragma unroll
        for (int it = 0; it < 2; ++it) {
            int rr = lr + it * 32;
            async_cp16(&Vs[buf][rr][lc],
                       Vt + ((size_t)bh * 64 + rr) * ALK + kt + lc);
        }
#else
        v8bf tk[2], tv[2];
#pragma unroll
        for (int it = 0; it < 2; ++it)
            tk[it] = *(const v8bf*)(Kb + ((size_t)bh * ALK + kt + lr + it * 32) * 64 + lc);
#pragma unroll
        for (int it = 0; it < 2; ++it)
            tv[it] = *(const v8bf*)(Vt + ((size_t)bh * 64 + lr + it * 32) * ALK + kt + lc);
#pragma unroll
        for (int it = 0; it < 2; ++it) *(v8bf*)(&Ks[buf][lr + it * 32][lc]) = tk[it];
#pragma unroll
        for (int it = 0; it < 2; ++it) *(v8bf*)(&Vs[buf][lr + it * 32][lc]) = tv[it];
#endif
    };

    // Q fragments held in registers for the whole sweep
    const bf16* qptr = Q + ((size_t)bh * ALQ + qrow0) * 64;
    v16bf aq[2];
    aq[0] = load_frag_a(qptr, 64, lane);       // d = 0..31
    aq[1] = load_frag_a(qptr + 32, 64, lane);  // d = 32..63

    float qmrow[8];
#pragma unroll
    for (int j = 0; j < 8; ++j)
        qmrow[j] = (float)qmask[b * ALQ + qrow0 + j + hlf * 8];

    float rowmax[8], rowsum[8];
#pragma unroll
    for (int j = 0; j < 8; ++j) { rowmax[j] = -3.0e38f; rowsum[j] = 0.0f; }
    v8f acc[4] = {};

    stageKV(0, 0);
    async_wait0();
    __syncthreads();

    const int nchunks = ALK / AKC;
    for (int cidx = 0; cidx < nchunks; ++cidx) {
        const int buf = cidx & 1;
        const int kt = cidx * AKC;
        if (cidx + 1 < nchunks) stageKV(buf ^ 1, kt + AKC);

        // S = Q K^T for this wave's 16 rows x 64 cols
        v8f s[4];
#pragma unroll
        for (int ni = 0; ni < 4; ++ni) {
            v8f cf = {};
#pragma unroll
            for (int kk = 0; kk < 2; ++kk) {
                v16bf bk = load_frag_b(&Ks[buf][ni * 16][kk * 32], GLD, lane);
                cf = wmma_bf16(aq[kk], bk, cf);
            }
            s[ni] = cf;
        }

        // mask + online softmax
        float fmv[4];
#pragma unroll
        for (int ni = 0; ni < 4; ++ni)
            fmv[ni] = (float)fmask[b * ALK + kt + ni * 16 + r];
#pragma unroll
        for (int j = 0; j < 8; ++j) {
            float mx = -3.0e38f;
#pragma unroll
            for (int ni = 0; ni < 4; ++ni) {
                float val = s[ni][j] + (1.0f - qmrow[j] * fmv[ni]) * NEGC;
                s[ni][j] = val;
                mx = fmaxf(mx, val);
            }
#pragma unroll
            for (int d = 1; d < 16; d <<= 1) mx = fmaxf(mx, __shfl_xor(mx, d, 32));
            float mnew = fmaxf(rowmax[j], mx);
            float corr = __expf(rowmax[j] - mnew);
            rowmax[j] = mnew;
            float rsum = 0.0f;
#pragma unroll
            for (int ni = 0; ni < 4; ++ni) {
                float p = __expf(s[ni][j] - mnew);
                s[ni][j] = p;
                rsum += p;
            }
#pragma unroll
            for (int d = 1; d < 16; d <<= 1) rsum += __shfl_xor(rsum, d, 32);
            rowsum[j] = rowsum[j] * corr + rsum;
#pragma unroll
            for (int ni = 0; ni < 4; ++ni) acc[ni][j] *= corr;
        }

        // C-layout P -> A-layout via wave-private LDS scratch
#pragma unroll
        for (int ni = 0; ni < 4; ++ni)
#pragma unroll
            for (int j = 0; j < 8; ++j)
                Ps[wave][j + hlf * 8][ni * 16 + r] = (bf16)s[ni][j];

        v16bf ap0 = load_frag_a(&Ps[wave][0][0], GLD, lane);
        v16bf ap1 = load_frag_a(&Ps[wave][0][32], GLD, lane);

        // O += P V
#pragma unroll
        for (int ni = 0; ni < 4; ++ni) {
            v16bf bv0 = load_frag_b(&Vs[buf][ni * 16][0], GLD, lane);
            v16bf bv1 = load_frag_b(&Vs[buf][ni * 16][32], GLD, lane);
            acc[ni] = wmma_bf16(ap0, bv0, acc[ni]);
            acc[ni] = wmma_bf16(ap1, bv1, acc[ni]);
        }
        async_wait0();
        __syncthreads();
    }

    // normalize and store (B, Lq, 1024) bf16
#pragma unroll
    for (int j = 0; j < 8; ++j) {
        float inv = 1.0f / rowsum[j];
        int q = qrow0 + j + hlf * 8;
        bf16* orow = O + ((size_t)b * ALQ + q) * 1024 + h * 64;
#pragma unroll
        for (int ni = 0; ni < 4; ++ni)
            orow[ni * 16 + r] = (bf16)(acc[ni][j] * inv);
    }
}

// ---------------------------------------------------------------------------
extern "C" void kernel_launch(void* const* d_in, const int* in_sizes, int n_in,
                              void* d_out, int out_size, void* d_ws, size_t ws_size,
                              hipStream_t stream) {
    const float* query = (const float*)d_in[0];
    const float* feat  = (const float*)d_in[1];
    const int* qmask   = (const int*)d_in[2];
    const int* fmask   = (const int*)d_in[3];
    const float* qpe   = (const float*)d_in[4];
    const float* fpe   = (const float*)d_in[5];
    const float* w_in  = (const float*)d_in[6];
    const float* b_in  = (const float*)d_in[7];
    const float* w_out = (const float*)d_in[8];
    const float* b_out = (const float*)d_in[9];
    float* out = (float*)d_out;

    char* ws = (char*)d_ws;
    size_t off = 0;
    auto alloc = [&](size_t bytes) -> char* {
        char* p = ws + off;
        off += (bytes + 255) & ~(size_t)255;
        return p;
    };
    bf16* q_in  = (bf16*)alloc((size_t)2048 * 1024 * 2);   // query + pe, bf16
    bf16* f_in  = (bf16*)alloc((size_t)16384 * 1024 * 2);  // feat  + pe, bf16
    bf16* wqkv  = (bf16*)alloc((size_t)3072 * 1024 * 2);
    bf16* wout  = (bf16*)alloc((size_t)1024 * 1024 * 2);
    bf16* qbuf  = (bf16*)alloc((size_t)2048 * 1024 * 2);   // (B,H,Lq,64)
    bf16* kbuf  = (bf16*)alloc((size_t)16384 * 1024 * 2);  // (B,H,Lk,64)
    bf16* vtbuf = (bf16*)alloc((size_t)16384 * 1024 * 2);  // (B,H,64,Lk)
    bf16* aout  = (bf16*)alloc((size_t)2048 * 1024 * 2);   // (B,Lq,1024)

    // prep
    add_pe_kernel<<<2048, 256, 0, stream>>>(query, qpe, q_in, 512 * 1024,
                                            2048 * 1024 / 4);
    add_pe_kernel<<<16384, 256, 0, stream>>>(feat, fpe, f_in, 4096 * 1024,
                                             16384 * 1024 / 4);
    cvt_bf16_kernel<<<3072, 256, 0, stream>>>(w_in, wqkv, 3072 * 1024 / 4);
    cvt_bf16_kernel<<<1024, 256, 0, stream>>>(w_out, wout, 1024 * 1024 / 4);

    // projections (q pre-scaled by 1/sqrt(64))
    gemm_bf16_kernel<<<dim3(2048 / GBM, 1024 / GBN), 256, 0, stream>>>(
        q_in, wqkv, b_in, qbuf, 2048, 1024, 1024, 0, 512, 0.125f);
    gemm_bf16_kernel<<<dim3(16384 / GBM, 1024 / GBN), 256, 0, stream>>>(
        f_in, wqkv + (size_t)1024 * 1024, b_in + 1024, kbuf, 16384, 1024, 1024, 0,
        4096, 1.0f);
    gemm_bf16_kernel<<<dim3(16384 / GBM, 1024 / GBN), 256, 0, stream>>>(
        f_in, wqkv + (size_t)2048 * 1024, b_in + 2048, vtbuf, 16384, 1024, 1024, 1,
        4096, 1.0f);

    // attention
    attn_kernel<<<dim3(64, 4), 256, 0, stream>>>(qbuf, kbuf, vtbuf, qmask, fmask,
                                                 aout);

    // output projection -> f32
    gemm_bf16_kernel<<<dim3(16, 16), 256, 0, stream>>>(
        aout, wout, b_out, out, 2048, 1024, 1024, 2, 512, 1.0f);
}
